// CausalSelfAttention_20014547599536
// MI455X (gfx1250) — matrix operand order repaired and checked by reading, not compiled, not run on hardware
//
#include <hip/hip_runtime.h>

typedef __bf16 bf16_t;
typedef __attribute__((ext_vector_type(16))) __bf16 v16bf;
typedef __attribute__((ext_vector_type(8)))  float  v8f;
typedef __attribute__((ext_vector_type(4)))  unsigned int v4u;
typedef __attribute__((ext_vector_type(4)))  float  v4f;
typedef __attribute__((ext_vector_type(4)))  unsigned int u32x4;
typedef __attribute__((ext_vector_type(8)))  int i32x8;
typedef __attribute__((ext_vector_type(4)))  int i32x4;

#if __has_builtin(__builtin_amdgcn_tensor_load_to_lds) && \
    __has_builtin(__builtin_amdgcn_s_wait_tensorcnt)
#define USE_TDM 1
#else
#define USE_TDM 0
#endif

union Frag {
  v16bf bf;
  v4u   u[2];
};

__device__ __forceinline__ v8f wmma_bf16(v16bf a, v16bf b, v8f c) {
  return __builtin_amdgcn_wmma_f32_16x16x32_bf16(false, a, false, b, (short)0, c,
                                                 false, false);
}

__device__ __forceinline__ v8f vzero8() {
  v8f z;
#pragma unroll
  for (int i = 0; i < 8; ++i) z[i] = 0.f;
  return z;
}

#define BATCH 2
#define SEQ   2048
#define CH    1024
#define NH    16
#define HD    64
#define BHT   (BATCH * NH)   // 32
#define MROWS (BATCH * SEQ)  // 4096

// ---------------------------------------------------------------------------
// Tiled GEMM: C[M,N] = A[M,K] * W[K,N] + bias, double-buffered LDS.
// MODE 0: A = fp32 x, scatter bf16 q/k/v in [B*H][T][64] layout (N = 3072)
// MODE 1: A = bf16 y, write fp32 out (N = 1024)
// Block = 256 threads (8 waves); 128x128 tile, BK = 32 (one bf16 WMMA step).
// ---------------------------------------------------------------------------
template <int MODE>
__global__ __launch_bounds__(256) void gemm_bf16_kernel(
    const void* __restrict__ Aptr, const float* __restrict__ W,
    const float* __restrict__ bias, bf16_t* __restrict__ qO,
    bf16_t* __restrict__ kO, bf16_t* __restrict__ vO,
    float* __restrict__ outF, int N, int K) {
  __shared__ bf16_t As[2][128][32];   // [buf][m][k]
  __shared__ bf16_t Bs[2][128][32];   // [buf][n][k] (transposed)

  const int tid  = threadIdx.x;
  const int wid  = tid >> 5;
  const int lane = tid & 31;
  const int half = lane >> 4;
  const int l15  = lane & 15;
  const int m0   = blockIdx.y * 128;
  const int n0   = blockIdx.x * 128;
  const int wm   = (wid >> 1) * 32;  // 4 waves along M
  const int wn   = (wid & 1) * 64;   // 2 waves along N

  const float*  Af = (const float*)Aptr;
  const bf16_t* Ah = (const bf16_t*)Aptr;

  v4f aF[4];  // MODE 0 staging regs
  v4u aH[2];  // MODE 1 staging regs
  v4f bF[4];

  auto loadRegs = [&](int kt) {
    if (MODE == 0) {
#pragma unroll
      for (int i = 0; i < 4; ++i) {
        int idx = tid + i * 256, row = idx >> 3, c4 = (idx & 7) * 4;
        aF[i] = *(const v4f*)(Af + (size_t)(m0 + row) * K + kt + c4);
      }
    } else {
#pragma unroll
      for (int i = 0; i < 2; ++i) {
        int idx = tid + i * 256, row = idx >> 2, c8 = (idx & 3) * 8;
        aH[i] = *(const v4u*)(Ah + (size_t)(m0 + row) * K + kt + c8);
      }
    }
#pragma unroll
    for (int i = 0; i < 4; ++i) {
      int idx = tid + i * 256, kr = idx >> 5, nc = (idx & 31) * 4;
      bF[i] = *(const v4f*)(W + (size_t)(kt + kr) * N + n0 + nc);
    }
  };
  auto storeRegs = [&](int p) {
    if (MODE == 0) {
#pragma unroll
      for (int i = 0; i < 4; ++i) {
        int idx = tid + i * 256, row = idx >> 3, c4 = (idx & 7) * 4;
        As[p][row][c4 + 0] = (bf16_t)aF[i].x;
        As[p][row][c4 + 1] = (bf16_t)aF[i].y;
        As[p][row][c4 + 2] = (bf16_t)aF[i].z;
        As[p][row][c4 + 3] = (bf16_t)aF[i].w;
      }
    } else {
#pragma unroll
      for (int i = 0; i < 2; ++i) {
        int idx = tid + i * 256, row = idx >> 2, c8 = (idx & 3) * 8;
        *(v4u*)&As[p][row][c8] = aH[i];
      }
    }
#pragma unroll
    for (int i = 0; i < 4; ++i) {
      int idx = tid + i * 256, kr = idx >> 5, nc = (idx & 31) * 4;
      Bs[p][nc + 0][kr] = (bf16_t)bF[i].x;
      Bs[p][nc + 1][kr] = (bf16_t)bF[i].y;
      Bs[p][nc + 2][kr] = (bf16_t)bF[i].z;
      Bs[p][nc + 3][kr] = (bf16_t)bF[i].w;
    }
  };

  v8f acc[2][4];
#pragma unroll
  for (int i = 0; i < 2; ++i)
#pragma unroll
    for (int j = 0; j < 4; ++j) acc[i][j] = vzero8();

  loadRegs(0);
  storeRegs(0);

  for (int kt = 0; kt < K; kt += 32) {
    const int p = (kt >> 5) & 1;
    __syncthreads();  // buf p fully staged; prev iter's reads of buf p^1 done
    const bool more = (kt + 32) < K;
    if (more) loadRegs(kt + 32);  // global loads in flight during WMMAs

    Frag af[2];
#pragma unroll
    for (int i = 0; i < 2; ++i) {
      int row = wm + i * 16 + l15;
      int ka  = half * 8;  // lanes<16: K 0..7 & 16..23; >=16: 8..15 & 24..31
      af[i].u[0] = *(const v4u*)&As[p][row][ka];
      af[i].u[1] = *(const v4u*)&As[p][row][ka + 16];
    }
    Frag bfm[4];
#pragma unroll
    for (int j = 0; j < 4; ++j) {
      int col = wn + j * 16 + l15;
      int kb  = half * 8;  // lanes<16: K 0..7 & 16..23; >=16: 8..15 & 24..31
      bfm[j].u[0] = *(const v4u*)&Bs[p][col][kb];
      bfm[j].u[1] = *(const v4u*)&Bs[p][col][kb + 16];
    }
#pragma unroll
    for (int i = 0; i < 2; ++i)
#pragma unroll
      for (int j = 0; j < 4; ++j)
        acc[i][j] = wmma_bf16(af[i].bf, bfm[j].bf, acc[i][j]);

    if (more) storeRegs(p ^ 1);
  }

  // ---- epilogue: D layout m = r + 8*half, n = l15 ----
#pragma unroll
  for (int i = 0; i < 2; ++i) {
#pragma unroll
    for (int j = 0; j < 4; ++j) {
      int gn = n0 + wn + j * 16 + l15;
      float bv = bias[gn];
#pragma unroll
      for (int r = 0; r < 8; ++r) {
        int gm = m0 + wm + i * 16 + r + 8 * half;  // token index
        float val = acc[i][j][r] + bv;
        if (MODE == 0) {
          int sec = gn >> 10;  // 0=q 1=k 2=v
          int c   = gn & 1023;
          int h   = c >> 6;
          int d   = c & 63;
          int b   = gm >> 11;
          int t   = gm & 2047;
          bf16_t* dst = (sec == 0) ? qO : (sec == 1) ? kO : vO;
          dst[(((size_t)(b * NH + h)) * SEQ + t) * HD + d] = (bf16_t)val;
        } else {
          outF[(size_t)gm * N + gn] = val;
        }
      }
    }
  }
}

// ---------------------------------------------------------------------------
// TDM: issue a 2D tensor_load_to_lds of a 64(keys) x 64(d) bf16 tile.
// Tensor: [2048 keys][64 d], dim0 (contiguous) = d, dim0_stride = 64.
// LDS receives rows (keys) contiguously -> matches Ks[key][d].
// ---------------------------------------------------------------------------
#if USE_TDM
__device__ __forceinline__ void tdm_load_k_tile(const bf16_t* gtile,
                                                unsigned lds_off) {
  unsigned long long ga = (unsigned long long)(uintptr_t)gtile;
  u32x4 g0;
  g0[0] = 1u;                                   // count = 1 (valid descriptor)
  g0[1] = lds_off;                              // lds_addr
  g0[2] = (unsigned)(ga & 0xffffffffu);         // global_addr[31:0]
  g0[3] = (unsigned)((ga >> 32) & 0x1ffffffu)   // global_addr[56:32]
          | 0x80000000u;                        // type = 2 ("image")
  i32x8 g1;
  g1[0] = 0x00010000;                           // data_size = 1 (2 bytes)
  g1[1] = (int)(64u << 16);                     // tensor_dim0 = 64 (lo16)
  g1[2] = (int)(2048u << 16);                   // tensor_dim1 = 2048 (lo16)
  g1[3] = (int)(64u << 16);                     // tile_dim0 = 64
  g1[4] = 64;                                   // tile_dim1 = 64, tile_dim2 = 0
  g1[5] = 64;                                   // tensor_dim0_stride = 64
  g1[6] = 0;
  g1[7] = 0;
  i32x4 z4;
  z4[0] = 0; z4[1] = 0; z4[2] = 0; z4[3] = 0;
  i32x8 z8;
#pragma unroll
  for (int i = 0; i < 8; ++i) z8[i] = 0;
  // clang-23 / therock arity: (g0, g1, g2, g3, g4, cpol)
  __builtin_amdgcn_tensor_load_to_lds(g0, g1, z4, z4, z8, 0);
}
#endif

// ---------------------------------------------------------------------------
// Flash attention: one (b,h) x 128 q-rows per workgroup; wave owns 16 q-rows.
// K-tiles of 64 keys staged via TDM (double-buffered); V transposed via VGPRs;
// online softmax in f32; P reshaped via per-wave LDS scratch.
// ---------------------------------------------------------------------------
__global__ __launch_bounds__(256) void flash_attn_kernel(
    const bf16_t* __restrict__ Q, const bf16_t* __restrict__ Kc,
    const bf16_t* __restrict__ V, bf16_t* __restrict__ Y) {
  __shared__ bf16_t Ks[2][64][64];   // [buf][key][d]
  __shared__ bf16_t Vt[64][64];      // transposed: [d][key]
  __shared__ bf16_t Pw[8][16][64];   // per-wave P scratch [m][key]

  const int tid   = threadIdx.x;
  const int wid   = tid >> 5;
  const int lane  = tid & 31;
  const int half  = lane >> 4;
  const int l15   = lane & 15;
  const int bh    = blockIdx.y;
  const int qRow0 = blockIdx.x * 128 + wid * 16;

  const bf16_t* qb = Q + (size_t)bh * SEQ * HD;
  const bf16_t* kb = Kc + (size_t)bh * SEQ * HD;
  const bf16_t* vb = V + (size_t)bh * SEQ * HD;

  // Q fragments (loaded once): A-layout, contraction over d (64 = 2 steps)
  Frag qa[2];
  {
    const bf16_t* base = qb + (size_t)(qRow0 + l15) * HD;
    qa[0].u[0] = *(const v4u*)(base + half * 8);
    qa[0].u[1] = *(const v4u*)(base + half * 8 + 16);
    qa[1].u[0] = *(const v4u*)(base + 32 + half * 8);
    qa[1].u[1] = *(const v4u*)(base + 32 + half * 8 + 16);
  }

  v8f o[4];
#pragma unroll
  for (int t = 0; t < 4; ++t) o[t] = vzero8();
  float rmax[8], rsum[8];
#pragma unroll
  for (int r = 0; r < 8; ++r) { rmax[r] = -__builtin_inff(); rsum[r] = 0.f; }

  const int kTiles = (blockIdx.x + 1) * 2;

#if USE_TDM
  const unsigned ksOff0 = (unsigned)(uintptr_t)&Ks[0][0][0];
  const unsigned ksOff1 = (unsigned)(uintptr_t)&Ks[1][0][0];
  if (wid == 0) tdm_load_k_tile(kb, ksOff0);  // prologue: tile 0 -> buf 0
#endif

  for (int kt = 0; kt < kTiles; ++kt) {
    const int k0 = kt * 64;
    const int p  = kt & 1;
    __syncthreads();  // prev iter's LDS reads done; Vt free to overwrite

    // ---- stage V transposed (all threads); 512 16B chunks ----
#pragma unroll
    for (int i = 0; i < 2; ++i) {
      int chunk = tid + i * 256;
      int key   = chunk >> 3;
      int d8    = (chunk & 7) * 8;
      v4u raw = *(const v4u*)(vb + (size_t)(k0 + key) * HD + d8);
      const bf16_t* e = (const bf16_t*)&raw;
#pragma unroll
      for (int j2 = 0; j2 < 8; ++j2) Vt[d8 + j2][key] = e[j2];
    }
#if !USE_TDM
    // fallback: stage K tile kt manually into Ks[p]
#pragma unroll
    for (int i = 0; i < 2; ++i) {
      int chunk = tid + i * 256;
      int key   = chunk >> 3;
      int d8    = (chunk & 7) * 8;
      *(v4u*)&Ks[p][key][d8] =
          *(const v4u*)(kb + (size_t)(k0 + key) * HD + d8);
    }
#else
    if (wid == 0) {
      if (kt + 1 < kTiles) {
        // issue next K tile into the other buffer, then retire current tile
        tdm_load_k_tile(kb + (size_t)(k0 + 64) * HD, p ? ksOff0 : ksOff1);
        __builtin_amdgcn_s_wait_tensorcnt(1);
      } else {
        __builtin_amdgcn_s_wait_tensorcnt(0);
      }
    }
#endif
    if (k0 + 64 < SEQ) {  // prefetch next V tile into cache
      __builtin_prefetch(vb + (size_t)(k0 + 64) * HD + lane * 16, 0, 1);
    }
    __syncthreads();  // Ks[p] (TDM) and Vt visible to all waves

    if (k0 <= qRow0 + 15) {  // causal: skip fully-masked tiles
      // ---- S = Q * K^T ----
      v8f s[4];
#pragma unroll
      for (int j = 0; j < 4; ++j) s[j] = vzero8();
#pragma unroll
      for (int c = 0; c < 2; ++c) {
        int kd = c * 32;
#pragma unroll
        for (int j = 0; j < 4; ++j) {
          Frag kf;
          int key = j * 16 + l15;
          kf.u[0] = *(const v4u*)&Ks[p][key][kd + half * 8];
          kf.u[1] = *(const v4u*)&Ks[p][key][kd + half * 8 + 16];
          s[j] = wmma_bf16(qa[c].bf, kf.bf, s[j]);
        }
      }
      // ---- scale + causal mask + tile row-max ----
      float tmax[8];
#pragma unroll
      for (int r = 0; r < 8; ++r) tmax[r] = -3.0e38f;
#pragma unroll
      for (int j = 0; j < 4; ++j) {
        int kg = k0 + j * 16 + l15;
#pragma unroll
        for (int r = 0; r < 8; ++r) {
          int qg = qRow0 + r + 8 * half;
          float sv = s[j][r] * 0.125f;  // 1/sqrt(64)
          sv = (kg > qg) ? -3.0e38f : sv;
          s[j][r] = sv;
          tmax[r] = fmaxf(tmax[r], sv);
        }
      }
#pragma unroll
      for (int md = 8; md >= 1; md >>= 1)
#pragma unroll
        for (int r = 0; r < 8; ++r)
          tmax[r] = fmaxf(tmax[r], __shfl_xor(tmax[r], md, 32));

      // ---- online rescale ----
#pragma unroll
      for (int r = 0; r < 8; ++r) {
        float nm    = fmaxf(rmax[r], tmax[r]);
        float alpha = __expf(rmax[r] - nm);
        rmax[r] = nm;
        rsum[r] *= alpha;
#pragma unroll
        for (int t = 0; t < 4; ++t) o[t][r] *= alpha;
      }
      // ---- P = exp(S - max), tile row-sum ----
      float tsum[8];
#pragma unroll
      for (int r = 0; r < 8; ++r) tsum[r] = 0.f;
#pragma unroll
      for (int j = 0; j < 4; ++j)
#pragma unroll
        for (int r = 0; r < 8; ++r) {
          float pv = __expf(s[j][r] - rmax[r]);
          s[j][r] = pv;
          tsum[r] += pv;
        }
#pragma unroll
      for (int md = 8; md >= 1; md >>= 1)
#pragma unroll
        for (int r = 0; r < 8; ++r) tsum[r] += __shfl_xor(tsum[r], md, 32);
#pragma unroll
      for (int r = 0; r < 8; ++r) rsum[r] += tsum[r];

      // ---- P (D-layout) -> LDS -> A-fragments ----
#pragma unroll
      for (int j = 0; j < 4; ++j)
#pragma unroll
        for (int r = 0; r < 8; ++r)
          Pw[wid][r + 8 * half][j * 16 + l15] = (bf16_t)s[j][r];
      asm volatile("s_wait_dscnt 0x0" ::: "memory");

      // ---- O += P * V ----
#pragma unroll
      for (int c = 0; c < 2; ++c) {
        int kk = c * 32;
        Frag pf;
        pf.u[0] = *(const v4u*)&Pw[wid][l15][kk + half * 8];
        pf.u[1] = *(const v4u*)&Pw[wid][l15][kk + half * 8 + 16];
#pragma unroll
        for (int t = 0; t < 4; ++t) {
          Frag vf;
          int dc = t * 16 + l15;
          vf.u[0] = *(const v4u*)&Vt[dc][kk + half * 8];
          vf.u[1] = *(const v4u*)&Vt[dc][kk + half * 8 + 16];
          o[t] = wmma_bf16(pf.bf, vf.bf, o[t]);
        }
      }
    }
  }

  // ---- normalize + store y as bf16 [B][T][C] ----
  const int b = bh >> 4, h = bh & 15;
#pragma unroll
  for (int t = 0; t < 4; ++t) {
#pragma unroll
    for (int r = 0; r < 8; ++r) {
      int qg = qRow0 + r + 8 * half;
      int d  = t * 16 + l15;
      float val = o[t][r] / rsum[r];
      Y[((size_t)(b * SEQ + qg)) * CH + h * HD + d] = (bf16_t)val;
    }
  }
}

// ---------------------------------------------------------------------------
extern "C" void kernel_launch(void* const* d_in, const int* in_sizes, int n_in,
                              void* d_out, int out_size, void* d_ws,
                              size_t ws_size, hipStream_t stream) {
  (void)in_sizes; (void)n_in; (void)out_size; (void)ws_size;
  const float* x  = (const float*)d_in[0];
  const float* Wa = (const float*)d_in[1];
  const float* ba = (const float*)d_in[2];
  const float* Wp = (const float*)d_in[3];
  const float* bp = (const float*)d_in[4];
  float* out = (float*)d_out;

  const size_t perBuf = (size_t)BHT * SEQ * HD;  // 4,194,304 bf16 elems (8 MB)
  bf16_t* q = (bf16_t*)d_ws;
  bf16_t* k = q + perBuf;
  bf16_t* v = k + perBuf;
  bf16_t* y = v + perBuf;  // [B][T][C] bf16

  // 1) qkv = x @ W_attn + b_attn  -> q/k/v bf16 [B*H][T][64]
  gemm_bf16_kernel<0><<<dim3((3 * CH) / 128, MROWS / 128), 256, 0, stream>>>(
      x, Wa, ba, q, k, v, nullptr, 3 * CH, CH);
  // 2) causal flash attention -> y bf16 [B][T][C]
  flash_attn_kernel<<<dim3(SEQ / 128, BHT), 256, 0, stream>>>(q, k, v, y);
  // 3) out = y @ W_proj + b_proj (fp32)
  gemm_bf16_kernel<1><<<dim3(CH / 128, MROWS / 128), 256, 0, stream>>>(
      y, Wp, bp, nullptr, nullptr, nullptr, out, CH, CH);
}
